// ContrastiveLoss_4672924418475
// MI455X (gfx1250) — matrix operand order repaired and checked by reading, not compiled, not run on hardware
//
#include <hip/hip_runtime.h>

#define CSPLIT 4      // column partitions for wave parallelism
#define CHUNK  64     // columns staged in LDS per TDM transfer
#define PADH   8      // f16 pad per row (16B) -> conflict-free DS reads

typedef __attribute__((ext_vector_type(16))) _Float16 v16h;
typedef __attribute__((ext_vector_type(8)))  _Float16 v8h;
typedef __attribute__((ext_vector_type(8)))  float    v8f;
typedef unsigned int u32;
typedef __attribute__((ext_vector_type(4))) u32 u32x4;
typedef __attribute__((ext_vector_type(8))) int i32x8;
typedef __attribute__((ext_vector_type(4))) int i32x4;

__device__ __forceinline__ v16h make_frag(v8h lo, v8h hi) {
  v16h r;
#pragma unroll
  for (int i = 0; i < 8; ++i) { r[i] = lo[i]; r[i + 8] = hi[i]; }
  return r;
}

// Low 32 bits of a flat shared-memory address == LDS byte offset (aperture
// mapping discards the upper bits).
__device__ __forceinline__ u32 lds_off32(const void* p) {
  return (u32)(unsigned long long)(uintptr_t)p;
}

// ---------------------------------------------------------------------------
// TDM: DMA one [CHUNK rows x 256 f16] tile (row-major) from global into LDS,
// inserting 16B of padding after every 512B row (pad_interval=128 DW code 6,
// pad_amount=4 DW code 3). Tracked by TENSORcnt. EXEC is ignored by TDM.
// ---------------------------------------------------------------------------
__device__ __forceinline__ void tdm_load_chunk(const _Float16* gsrc, u32 lds_byte_off) {
  const unsigned long long ga = (unsigned long long)(uintptr_t)gsrc;
  u32x4 g0;
  g0[0] = 1u;                                             // count=1 (valid user D#)
  g0[1] = lds_byte_off;                                   // lds_addr
  g0[2] = (u32)(ga & 0xffffffffu);                        // global_addr[31:0]
  g0[3] = (u32)((ga >> 32) & 0x01ffffffu) | 0x80000000u;  // addr[56:32] | type=2

  i32x8 g1;
  // data_size=2B(code1)<<16 | pad_enable<<20 | pad_interval(6)<<22 | pad_amount(3)<<25
  g1[0] = (1 << 16) | (1 << 20) | (6 << 22) | (3 << 25);
  g1[1] = (int)(256u << 16);                 // tensor_dim0[15:0]=256 in [31:16]
  g1[2] = (int)(8192u << 16);                // tensor_dim0[31:16]=0 | tensor_dim1[15:0]=8192
  g1[3] = (int)(256u << 16);                 // tensor_dim1[31:16]=0 | tile_dim0=256
  g1[4] = CHUNK;                             // tile_dim1=CHUNK | tile_dim2=0
  g1[5] = 256;                               // tensor_dim0_stride[31:0]=256 elements
  g1[6] = 0;                                 // stride0[47:32]=0 | dim1_stride[15:0]=0
  g1[7] = 0;

  i32x4 z4 = {0, 0, 0, 0};
  i32x8 z8 = {0, 0, 0, 0, 0, 0, 0, 0};
  __builtin_amdgcn_tensor_load_to_lds(g0, g1, z4, z4, z8, 0);
}

// ---------------------------------------------------------------------------
// Kernel 1: L2-normalize rows of [N,256] f32 -> f16 workspace rows.
// ---------------------------------------------------------------------------
__global__ __launch_bounds__(256) void fnorm_kernel(const float* __restrict__ feat,
                                                    _Float16* __restrict__ out, int N) {
  const int lane = threadIdx.x & 31;
  const int wave = threadIdx.x >> 5;
  const int row  = blockIdx.x * 8 + wave;
  const float* rp = feat + (size_t)row * 256 + lane * 8;
  float x[8];
  *(float4*)(x)     = *(const float4*)(rp);
  *(float4*)(x + 4) = *(const float4*)(rp + 4);
  float ss = 0.f;
#pragma unroll
  for (int i = 0; i < 8; ++i) ss += x[i] * x[i];
#pragma unroll
  for (int m = 1; m < 32; m <<= 1) ss += __shfl_xor(ss, m, 32);
  const float inv = rsqrtf(fmaxf(ss, 1e-24f));
  v8h h;
#pragma unroll
  for (int i = 0; i < 8; ++i) h[i] = (_Float16)(x[i] * inv);
  *(v8h*)(out + (size_t)row * 256 + lane * 8) = h;
}

// ---------------------------------------------------------------------------
// Kernel 2: fused sim/exp/mask/row-sum. Block = 8 waves = 128 rows.
// B columns staged through LDS by TDM (double buffered), shared by all waves.
// ---------------------------------------------------------------------------
__global__ __launch_bounds__(256) void simtile_kernel(
    const _Float16* __restrict__ f, const long long* __restrict__ labels,
    float* __restrict__ pos_part, float* __restrict__ tot_part, int N) {
  __shared__ _Float16 buf[2][CHUNK][256 + PADH];

  const int lane = threadIdx.x & 31;
  const int wave = threadIdx.x >> 5;
  const int nl   = lane & 15;
  const int sel  = lane >> 4;
  const int row0 = blockIdx.x * 128 + wave * 16;
  const int cwidth = N / CSPLIT;
  const int cbase  = blockIdx.y * cwidth;
  const int nch    = cwidth / CHUNK;
  const bool issuer = (threadIdx.x < 32);   // wave 0 drives the TDM

  // A fragments for full K=256 held in registers.
  const _Float16* aptr = f + (size_t)(row0 + nl) * 256;
  v16h a[8];
#pragma unroll
  for (int kf = 0; kf < 8; ++kf) {
    const int kb = kf * 32 + sel * 8;
    a[kf] = make_frag(*(const v8h*)(aptr + kb), *(const v8h*)(aptr + kb + 16));
  }
  int rl[8];
#pragma unroll
  for (int v = 0; v < 8; ++v) rl[v] = (int)labels[row0 + sel * 8 + v];

  float psum[8], tsum[8];
#pragma unroll
  for (int v = 0; v < 8; ++v) { psum[v] = 0.f; tsum[v] = 0.f; }

  // Preload chunk 0, then double-buffer: issue i+1, compute i, wait, barrier.
  if (issuer) {
    tdm_load_chunk(f + (size_t)cbase * 256, lds_off32(&buf[0][0][0]));
    __builtin_amdgcn_s_wait_tensorcnt(0);
  }
  __syncthreads();

  for (int ch = 0; ch < nch; ++ch) {
    const int c0g = cbase + ch * CHUNK;
    if (issuer && (ch + 1 < nch))
      tdm_load_chunk(f + (size_t)(c0g + CHUNK) * 256,
                     lds_off32(&buf[(ch + 1) & 1][0][0]));

    const _Float16* lbuf = &buf[ch & 1][0][0];
#pragma unroll
    for (int t = 0; t < CHUNK; t += 16) {
      const int cl = (int)labels[c0g + t + nl];
      const _Float16* bp = lbuf + (size_t)(t + nl) * (256 + PADH);
      v16h b[8];
#pragma unroll
      for (int kf = 0; kf < 8; ++kf) {
        const int kb = kf * 32 + sel * 8;    // B: two 8-K runs per half-wave (table order, matches A)
        b[kf] = make_frag(*(const v8h*)(bp + kb), *(const v8h*)(bp + kb + 16));
      }
      v8f acc = {};
#pragma unroll
      for (int kf = 0; kf < 8; ++kf)
        acc = __builtin_amdgcn_wmma_f32_16x16x32_f16(false, a[kf], false, b[kf],
                                                     (short)0, acc, false, false);
#pragma unroll
      for (int v = 0; v < 8; ++v) {
        const float e = __expf(acc[v] * 10.0f);   // 1/TEMPERATURE
        tsum[v] += e;
        psum[v] += (rl[v] == cl) ? e : 0.f;
      }
    }

    if (issuer) __builtin_amdgcn_s_wait_tensorcnt(0);
    __syncthreads();
  }

  // Reduce across the 16 lanes of each half-wave.
#pragma unroll
  for (int v = 0; v < 8; ++v) {
#pragma unroll
    for (int m = 1; m < 16; m <<= 1) {
      psum[v] += __shfl_xor(psum[v], m, 32);
      tsum[v] += __shfl_xor(tsum[v], m, 32);
    }
  }
  if (nl == 0) {
    const size_t base = (size_t)blockIdx.y * N + row0 + sel * 8;
#pragma unroll
    for (int v = 0; v < 8; ++v) {
      pos_part[base + v] = psum[v];
      tot_part[base + v] = tsum[v];
    }
  }
}

// ---------------------------------------------------------------------------
// Kernel 3: combine column partials -> per-row loss -> mean (deterministic).
// ---------------------------------------------------------------------------
__global__ __launch_bounds__(1024) void finalize_kernel(
    const float* __restrict__ pos_part, const float* __restrict__ tot_part,
    float* __restrict__ out, int N) {
  __shared__ float sm[1024];
  float s = 0.f;
  for (int r = threadIdx.x; r < N; r += 1024) {
    float p = 0.f, t = 0.f;
#pragma unroll
    for (int c = 0; c < CSPLIT; ++c) {
      p += pos_part[(size_t)c * N + r];
      t += tot_part[(size_t)c * N + r];
    }
    s += __logf(t) - __logf(p);
  }
  sm[threadIdx.x] = s;
  __syncthreads();
  for (int off = 512; off > 0; off >>= 1) {
    if ((int)threadIdx.x < off) sm[threadIdx.x] += sm[threadIdx.x + off];
    __syncthreads();
  }
  if (threadIdx.x == 0) out[0] = sm[0] / (float)N;
}

// ---------------------------------------------------------------------------
extern "C" void kernel_launch(void* const* d_in, const int* in_sizes, int n_in,
                              void* d_out, int out_size, void* d_ws, size_t ws_size,
                              hipStream_t stream) {
  const float*     feat   = (const float*)d_in[0];
  const long long* labels = (const long long*)d_in[1];  // int64 labels
  const int N = in_sizes[1];            // 8192
  const int D = 256;

  _Float16* f16f = (_Float16*)d_ws;
  float* pos_part = (float*)((char*)d_ws + (size_t)N * D * sizeof(_Float16));
  float* tot_part = pos_part + (size_t)CSPLIT * N;

  fnorm_kernel<<<N / 8, 256, 0, stream>>>(feat, f16f, N);

  dim3 grid(N / 128, CSPLIT);
  simtile_kernel<<<grid, 256, 0, stream>>>(f16f, labels, pos_part, tot_part, N);

  finalize_kernel<<<1, 1024, 0, stream>>>(pos_part, tot_part, (float*)d_out, N);
}
